// MambaLayer_49392123904378
// MI455X (gfx1250) — matrix operand order repaired and checked by reading, not compiled, not run on hardware
//
#include <hip/hip_runtime.h>
#include <hip/hip_bf16.h>
#include <math.h>

typedef __bf16 bf16_t;
typedef __attribute__((ext_vector_type(16))) __bf16 v16bf;
typedef __attribute__((ext_vector_type(8)))  __bf16 v8bf;
typedef __attribute__((ext_vector_type(8)))  float  v8f;
typedef __attribute__((ext_vector_type(4)))  int    v4i;

typedef __attribute__((address_space(1))) v4i v4i_g;   // global
typedef __attribute__((address_space(3))) v4i v4i_l;   // LDS

#define BB 2
#define LL 2048
#define DD 1024
#define EE 2048
#define NNS 16
#define KC 4
#define RR 64
#define MTOT (BB*LL)   // 4096 token rows

// Tiled-GEMM geometry
#define TBM 128
#define TBN 128
#define TBK 64
#define TBKP 72   // padded LDS row stride (144B, odd multiple of 16B)

#if defined(__gfx1250__) && \
    __has_builtin(__builtin_amdgcn_global_load_async_to_lds_b128) && \
    __has_builtin(__builtin_amdgcn_s_wait_asynccnt)
#define USE_ASYNC 1
#else
#define USE_ASYNC 0
#endif

// ---------------------------------------------------------------- convert
__global__ void f32_to_bf16_kernel(const float* __restrict__ in,
                                   bf16_t* __restrict__ out, int n) {
  int i = blockIdx.x * blockDim.x + threadIdx.x;
  if (i < n) out[i] = (bf16_t)in[i];
}

// ------------------------------------------------- WMMA fragment loaders
// A fragment 16x32 bf16: lanes 0-15 row m hold K=0..15,
// lanes 16-31 row m hold K=16..31 (contiguous, matches frag_b); 8 VGPRs.
__device__ __forceinline__ v16bf frag_a(const bf16_t* base, int lds, int row16,
                                        int kk, int lane) {
  int h = (lane >> 4) & 1;
  const bf16_t* p = base + (size_t)(row16 + (lane & 15)) * lds + kk + h * 16;
  union { v16bf v; v8bf q[2]; } u;
  u.q[0] = *(const v8bf*)(p);
  u.q[1] = *(const v8bf*)(p + 8);
  return u.v;
}
// B fragment 32x16 bf16: lane n (0-15) holds col n K=0..15; lanes 16-31 hold
// col n K=16..31 (contiguous 32B per lane from row n of W / LDS tile row).
__device__ __forceinline__ v16bf frag_b(const bf16_t* base, int lds, int col16,
                                        int kk, int lane) {
  int h = (lane >> 4) & 1;
  return *(const v16bf*)(base + (size_t)(col16 + (lane & 15)) * lds + kk + h * 16);
}

__device__ __forceinline__ float softplus_f(float x) {
  return (x > 20.f) ? x : log1pf(__expf(x));
}

// 16B global -> LDS copy (async on CDNA5, tracked by ASYNCcnt)
__device__ __forceinline__ void cp_b128(const bf16_t* g, bf16_t* l) {
#if USE_ASYNC
  const v4i* gv = (const v4i*)g;
  v4i*       lv = (v4i*)l;
  __builtin_amdgcn_global_load_async_to_lds_b128(
      (v4i_g*)gv, (v4i_l*)lv, 0, 0);
#else
  *(v8bf*)l = *(const v8bf*)g;
#endif
}

#if USE_ASYNC
#define ASYNC_WAIT(n) __builtin_amdgcn_s_wait_asynccnt(n)
#else
#define ASYNC_WAIT(n)
#endif

// --------------------------------------------- LDS-tiled double-buffered GEMM
// C[M,N] = A[M,K](bf16 rowmajor) x B[N,K](bf16 rowmajor)^T, f32 accum.
// 256 threads = 8 waves; block tile 128x128; wave tile 32x64 (2x4 WMMA tiles).
__device__ __forceinline__ void copy_stage(bf16_t* sA, bf16_t* sB,
                                           const bf16_t* A, const bf16_t* Bw,
                                           int tm0, int tn0, int k0, int K) {
  int c = threadIdx.x;
  #pragma unroll
  for (int i = 0; i < (TBM * TBK / 8) / 256; ++i, c += 256) {
    int row = c >> 3;
    int col = (c & 7) * 8;
    cp_b128(A + (size_t)(tm0 + row) * K + k0 + col, sA + row * TBKP + col);
  }
  c = threadIdx.x;
  #pragma unroll
  for (int i = 0; i < (TBN * TBK / 8) / 256; ++i, c += 256) {
    int row = c >> 3;
    int col = (c & 7) * 8;
    cp_b128(Bw + (size_t)(tn0 + row) * K + k0 + col, sB + row * TBKP + col);
  }
}

template<int EPI>
__global__ __launch_bounds__(256)
void gemm_bf16_wmma_tiled_kernel(const bf16_t* __restrict__ A,
                                 const bf16_t* __restrict__ Bw,
                                 float* __restrict__ C,
                                 int M, int N, int K,
                                 const float* __restrict__ auxf) {
  __shared__ bf16_t sA[2][TBM * TBKP];
  __shared__ bf16_t sB[2][TBN * TBKP];

  const int lane = threadIdx.x & 31;
  const int wave = threadIdx.x >> 5;
  const int nTilesN = N / TBN;
  const int tm0 = (blockIdx.x / nTilesN) * TBM;
  const int tn0 = (blockIdx.x % nTilesN) * TBN;
  const int wm = (wave >> 1) * 32;   // wave row base within block tile
  const int wn = (wave & 1) * 64;    // wave col base within block tile

  v8f acc[2][4] = {};

  const int nk = K / TBK;
  copy_stage(sA[0], sB[0], A, Bw, tm0, tn0, 0, K);

  for (int ks = 0; ks < nk; ++ks) {
    const int cur = ks & 1;
    if (ks + 1 < nk) {
      copy_stage(sA[cur ^ 1], sB[cur ^ 1], A, Bw, tm0, tn0, (ks + 1) * TBK, K);
      ASYNC_WAIT(8);   // wait only for the *previous* stage's 8 copies/thread
    } else {
      ASYNC_WAIT(0);
    }
    __syncthreads();

    const bf16_t* As = sA[cur];
    const bf16_t* Bs = sB[cur];
    #pragma unroll
    for (int kk = 0; kk < TBK; kk += 32) {
      v16bf a0 = frag_a(As, TBKP, wm + 0,  kk, lane);
      v16bf a1 = frag_a(As, TBKP, wm + 16, kk, lane);
      #pragma unroll
      for (int j = 0; j < 4; ++j) {
        v16bf bf = frag_b(Bs, TBKP, wn + j * 16, kk, lane);
        acc[0][j] = __builtin_amdgcn_wmma_f32_16x16x32_bf16(
            false, a0, false, bf, (short)0, acc[0][j], false, false);
        acc[1][j] = __builtin_amdgcn_wmma_f32_16x16x32_bf16(
            false, a1, false, bf, (short)0, acc[1][j], false, false);
      }
    }
    __syncthreads();   // everyone done reading before buffer is overwritten
  }

  // C layout: VGPR r -> (M = base + 8*half + r, N = base + (lane&15))
  const int h  = (lane >> 4) & 1;
  const int nl = lane & 15;
  #pragma unroll
  for (int im = 0; im < 2; ++im) {
    #pragma unroll
    for (int j = 0; j < 4; ++j) {
      int ng = tn0 + wn + j * 16 + nl;
      #pragma unroll
      for (int r = 0; r < 8; ++r) {
        int mg = tm0 + wm + im * 16 + h * 8 + r;
        float v = acc[im][j][r];
        if (EPI == 3) v += auxf[(size_t)mg * N + ng];
        C[(size_t)mg * N + ng] = v;
      }
    }
  }
}

// ---------------------------------------- direct (small) WMMA GEMM, per-wave
// Used where LDS tiling buys nothing: dbl (N=96, B fits in L2) and
// delta (K=64, a single k-stage).
// EPI: 1 also store bf16 copy of cols<RR (dt), 2 softplus(+b_dt).
template<int NT, int EPI>
__global__ void gemm_bf16_wmma_kernel(const bf16_t* __restrict__ A,
                                      const bf16_t* __restrict__ Bw,
                                      float* __restrict__ C,
                                      int M, int N, int K,
                                      const float* __restrict__ auxf,
                                      bf16_t* __restrict__ auxb) {
  const int lane = threadIdx.x & 31;
  const int wave = threadIdx.x >> 5;
  const int stripsPerRow = N / (16 * NT);
  const int strip = blockIdx.x * (blockDim.x >> 5) + wave;
  if (strip >= (M / 16) * stripsPerRow) return;
  const int tm  = strip / stripsPerRow;
  const int tn0 = (strip % stripsPerRow) * NT;

  v8f acc[NT] = {};

  for (int k0 = 0; k0 < K; k0 += 32) {
    v16bf af = frag_a(A, K, tm * 16, k0, lane);
    if (k0 + 32 < K)
      __builtin_prefetch(A + (size_t)(tm * 16 + (lane & 15)) * K + k0 + 32, 0, 1);
    #pragma unroll
    for (int t = 0; t < NT; ++t) {
      v16bf bf = frag_b(Bw, K, (tn0 + t) * 16, k0, lane);
      acc[t] = __builtin_amdgcn_wmma_f32_16x16x32_bf16(
          false, af, false, bf, (short)0, acc[t], false, false);
    }
  }

  const int h  = (lane >> 4) & 1;
  const int nl = lane & 15;
  #pragma unroll
  for (int t = 0; t < NT; ++t) {
    int ng = (tn0 + t) * 16 + nl;
    #pragma unroll
    for (int r = 0; r < 8; ++r) {
      int mg = tm * 16 + h * 8 + r;
      float v = acc[t][r];
      if (EPI == 2) v = softplus_f(v + auxf[ng]);
      C[(size_t)mg * N + ng] = v;
      if (EPI == 1) { if (ng < RR) auxb[(size_t)mg * RR + ng] = (bf16_t)v; }
    }
  }
}

// ----------------------------------------------- depthwise causal conv+SiLU
__global__ void conv_silu_kernel(const float* __restrict__ xz,
                                 const float* __restrict__ Wc,
                                 const float* __restrict__ bc,
                                 float* __restrict__ u,
                                 bf16_t* __restrict__ ub) {
  size_t idx = (size_t)blockIdx.x * blockDim.x + threadIdx.x;
  if (idx >= (size_t)MTOT * EE) return;
  int e = (int)(idx % EE);
  size_t m = idx / EE;          // b*L + l
  int l = (int)(m % LL);
  float acc = bc[e];
  #pragma unroll
  for (int k = 0; k < KC; ++k) {
    int ll = l - (KC - 1) + k;
    if (ll >= 0)
      acc += xz[(m + (size_t)(ll - l)) * (size_t)(2 * EE) + e] * Wc[e * KC + k];
  }
  float s = acc / (1.f + __expf(-acc));   // silu
  u[m * (size_t)EE + e]  = s;
  ub[m * (size_t)EE + e] = (bf16_t)s;
}

// --------------------------------------- selective scan + skip + SiLU gate
__global__ void scan_gate_kernel(const float* __restrict__ delta,
                                 const float* __restrict__ u,
                                 const float* __restrict__ dbl,
                                 const float* __restrict__ xz,
                                 const float* __restrict__ A_log,
                                 const float* __restrict__ D_skip,
                                 bf16_t* __restrict__ yb) {
  int t = blockIdx.x * blockDim.x + threadIdx.x;
  if (t >= BB * EE) return;
  int b = t / EE;
  int e = t % EE;
  float Ae[NNS];
  #pragma unroll
  for (int n = 0; n < NNS; ++n) Ae[n] = -__expf(A_log[e * NNS + n]);
  float dsk = D_skip[e];
  float h[NNS];
  #pragma unroll
  for (int n = 0; n < NNS; ++n) h[n] = 0.f;

  for (int l = 0; l < LL; ++l) {
    size_t m = (size_t)b * LL + l;
    float d  = delta[m * EE + e];
    float ut = u[m * EE + e];
    float du = d * ut;
    const float* bc = dbl + m * 96;   // [0:64)=dt, [64:80)=B, [80:96)=C
    float yt = 0.f;
    #pragma unroll
    for (int n = 0; n < NNS; ++n) {
      h[n] = __expf(d * Ae[n]) * h[n] + du * bc[64 + n];
      yt  += h[n] * bc[80 + n];
    }
    float z = xz[m * (size_t)(2 * EE) + EE + e];
    float g = z / (1.f + __expf(-z));
    float y = (yt + ut * dsk) * g;
    yb[m * EE + e] = (bf16_t)y;
  }
}

// ------------------------------------------------------------- layernorm
__global__ void layernorm_kernel(const float* __restrict__ res,
                                 const float* __restrict__ w,
                                 const float* __restrict__ bia,
                                 float* __restrict__ out) {
  __shared__ float s1[256];
  __shared__ float s2[256];
  int m = blockIdx.x;
  int tid = threadIdx.x;
  const float* row = res + (size_t)m * DD;
  float vals[DD / 256];
  float sum = 0.f, sq = 0.f;
  #pragma unroll
  for (int i = 0; i < DD / 256; ++i) {
    float v = row[tid + i * 256];
    vals[i] = v; sum += v; sq += v * v;
  }
  s1[tid] = sum; s2[tid] = sq;
  __syncthreads();
  for (int off = 128; off > 0; off >>= 1) {
    if (tid < off) { s1[tid] += s1[tid + off]; s2[tid] += s2[tid + off]; }
    __syncthreads();
  }
  float mu   = s1[0] / DD;
  float var  = s2[0] / DD - mu * mu;
  float rstd = rsqrtf(var + 1e-5f);
  #pragma unroll
  for (int i = 0; i < DD / 256; ++i) {
    int d = tid + i * 256;
    out[(size_t)m * DD + d] = (vals[i] - mu) * rstd * w[d] + bia[d];
  }
}

// ---------------------------------------------------------------- launch
extern "C" void kernel_launch(void* const* d_in, const int* in_sizes, int n_in,
                              void* d_out, int out_size, void* d_ws, size_t ws_size,
                              hipStream_t stream) {
  const float* x      = (const float*)d_in[0];
  const float* W_in   = (const float*)d_in[1];
  const float* W_conv = (const float*)d_in[2];
  const float* b_conv = (const float*)d_in[3];
  const float* W_x    = (const float*)d_in[4];
  const float* W_dt   = (const float*)d_in[5];
  const float* b_dt   = (const float*)d_in[6];
  const float* A_log  = (const float*)d_in[7];
  const float* D_skip = (const float*)d_in[8];
  const float* W_out  = (const float*)d_in[9];
  const float* ln_w   = (const float*)d_in[10];
  const float* ln_b   = (const float*)d_in[11];
  float* out = (float*)d_out;

  char* ws = (char*)d_ws;
  size_t off = 0;
  auto take = [&](size_t bytes) -> char* {
    char* p = ws + off;
    off = (off + bytes + 255) & ~(size_t)255;
    return p;
  };
  bf16_t* Win_b  = (bf16_t*)take((size_t)2 * EE * DD * 2);
  bf16_t* Wx_b   = (bf16_t*)take((size_t)96 * EE * 2);
  bf16_t* Wdt_b  = (bf16_t*)take((size_t)EE * RR * 2);
  bf16_t* Wout_b = (bf16_t*)take((size_t)DD * EE * 2);
  bf16_t* x_b    = (bf16_t*)take((size_t)MTOT * DD * 2);
  float*  xzf    = (float*) take((size_t)MTOT * 2 * EE * 4);
  float*  uf     = (float*) take((size_t)MTOT * EE * 4);
  bf16_t* u_b    = (bf16_t*)take((size_t)MTOT * EE * 2);
  float*  dblf   = (float*) take((size_t)MTOT * 96 * 4);
  bf16_t* dt_b   = (bf16_t*)take((size_t)MTOT * RR * 2);
  float*  deltaf = (float*) take((size_t)MTOT * EE * 4);
  bf16_t* y_b    = (bf16_t*)take((size_t)MTOT * EE * 2);
  float*  resf   = (float*) take((size_t)MTOT * DD * 4);

  const int TPB = 256;
  auto cvt = [&](const float* src, bf16_t* dst, int n) {
    f32_to_bf16_kernel<<<(n + TPB - 1) / TPB, TPB, 0, stream>>>(src, dst, n);
  };
  cvt(W_in,  Win_b,  2 * EE * DD);
  cvt(W_x,   Wx_b,   96 * EE);
  cvt(W_dt,  Wdt_b,  EE * RR);
  cvt(W_out, Wout_b, DD * EE);
  cvt(x,     x_b,    MTOT * DD);

  // in-proj: xz = x · W_inᵀ   (M=4096, N=4096, K=1024) — LDS-tiled WMMA
  {
    int blocks = (MTOT / TBM) * ((2 * EE) / TBN);   // 1024
    gemm_bf16_wmma_tiled_kernel<0><<<blocks, 256, 0, stream>>>(
        x_b, Win_b, xzf, MTOT, 2 * EE, DD, nullptr);
  }
  // depthwise conv + silu -> u (f32 + bf16)
  {
    size_t n = (size_t)MTOT * EE;
    conv_silu_kernel<<<(int)((n + TPB - 1) / TPB), TPB, 0, stream>>>(
        xzf, W_conv, b_conv, uf, u_b);
  }
  // dbl = u · W_xᵀ  (M=4096, N=96, K=2048); side-store dt as bf16
  {
    int strips = (MTOT / 16) * (96 / (16 * 2));
    gemm_bf16_wmma_kernel<2, 1><<<strips / 8, 256, 0, stream>>>(
        u_b, Wx_b, dblf, MTOT, 96, EE, nullptr, dt_b);
  }
  // delta = softplus(dt · W_dtᵀ + b_dt)  (M=4096, N=2048, K=64)
  {
    int strips = (MTOT / 16) * (EE / (16 * 4));
    gemm_bf16_wmma_kernel<4, 2><<<strips / 8, 256, 0, stream>>>(
        dt_b, Wdt_b, deltaf, MTOT, EE, RR, b_dt, nullptr);
  }
  // selective scan + skip + gate -> y (bf16)
  scan_gate_kernel<<<(BB * EE) / TPB, TPB, 0, stream>>>(
      deltaf, uf, dblf, xzf, A_log, D_skip, y_b);
  // out-proj + residual: res = y · W_outᵀ + x  (M=4096, N=1024, K=2048) — tiled
  {
    int blocks = (MTOT / TBM) * (DD / TBN);         // 256
    gemm_bf16_wmma_tiled_kernel<3><<<blocks, 256, 0, stream>>>(
        y_b, Wout_b, resf, MTOT, DD, EE, x);
  }
  // layernorm
  layernorm_kernel<<<MTOT, 256, 0, stream>>>(resf, ln_w, ln_b, out);
}
